// ModernAttention_82781199663669
// MI455X (gfx1250) — compile-verified, matrix operand order checked by reading, not run on hardware
//
#include <hip/hip_runtime.h>

// ---------------------------------------------------------------------------
// MI455X (gfx1250, wave32) RoPE attention, f16 WMMA pipeline.
// ---------------------------------------------------------------------------

typedef __attribute__((ext_vector_type(16))) _Float16 v16h;
typedef __attribute__((ext_vector_type(8)))  float    v8f;

union Frag16 { v16h v; uint4 u[2]; };

#define D_MODEL   1024
#define NUM_HEADS 16
#define HEAD_DIM  64
#define BATCH     2
#define SEQ       2048
#define MROWS     (BATCH * SEQ)   // 4096

// ---------------------------------------------------------------------------
// fp32 -> fp16 elementwise convert
// ---------------------------------------------------------------------------
__global__ void cvt_f32_f16(const float* __restrict__ in, _Float16* __restrict__ out, int n) {
    int i = blockIdx.x * blockDim.x + threadIdx.x;
    if (i < n) out[i] = (_Float16)in[i];
}

// ---------------------------------------------------------------------------
// Generic GEMM: C[M][N] = A[M][K] * W[N][K]^T   (A,W f16 row-major, C f32)
// Block = 256 threads (8 waves). Tile 128(M) x 128(N), BK=32.
// Wave (wm,wn) in 4x2 grid computes a 32x64 sub-tile = 2x4 WMMA fragments.
// Per k-step per wave: 8 v_wmma against 12 ds_load_b128.
// ---------------------------------------------------------------------------
__global__ __launch_bounds__(256)
void gemm_nt_f16(const _Float16* __restrict__ A, const _Float16* __restrict__ W,
                 float* __restrict__ C, int M, int N, int K) {
    __shared__ __align__(16) _Float16 Atile[128][32];
    __shared__ __align__(16) _Float16 Btile[128][32];

    const int t    = threadIdx.x;
    const int wave = t >> 5, lane = t & 31;
    const int half = lane >> 4, l16 = lane & 15;
    const int wm   = wave >> 1, wn = wave & 1;
    const int bm   = blockIdx.y * 128, bn = blockIdx.x * 128;

    v8f acc[2][4] = {};

    const int ar = t >> 1, ac = (t & 1) * 16;   // both tiles: 128 rows x 32, 16 elems/thread

    for (int k0 = 0; k0 < K; k0 += 32) {
        __syncthreads();
        *(uint4*)&Atile[ar][ac]     = *(const uint4*)&A[(size_t)(bm + ar) * K + k0 + ac];
        *(uint4*)&Atile[ar][ac + 8] = *(const uint4*)&A[(size_t)(bm + ar) * K + k0 + ac + 8];
        *(uint4*)&Btile[ar][ac]     = *(const uint4*)&W[(size_t)(bn + ar) * K + k0 + ac];
        *(uint4*)&Btile[ar][ac + 8] = *(const uint4*)&W[(size_t)(bn + ar) * K + k0 + ac + 8];
        __syncthreads();

        Frag16 a[2], b[4];
#pragma unroll
        for (int i = 0; i < 2; ++i) {
            int r = wm * 32 + i * 16 + l16;
            a[i].u[0] = *(const uint4*)&Atile[r][half * 8];
            a[i].u[1] = *(const uint4*)&Atile[r][16 + half * 8];
        }
#pragma unroll
        for (int j = 0; j < 4; ++j) {
            int r = wn * 64 + j * 16 + l16;
            b[j].u[0] = *(const uint4*)&Btile[r][half * 8];
            b[j].u[1] = *(const uint4*)&Btile[r][16 + half * 8];
        }
#pragma unroll
        for (int i = 0; i < 2; ++i)
#pragma unroll
            for (int j = 0; j < 4; ++j)
                acc[i][j] = __builtin_amdgcn_wmma_f32_16x16x32_f16(
                    false, a[i].v, false, b[j].v, (short)0, acc[i][j], false, false);
    }

#pragma unroll
    for (int i = 0; i < 2; ++i)
#pragma unroll
        for (int j = 0; j < 4; ++j) {
            int col = bn + wn * 64 + j * 16 + l16;
#pragma unroll
            for (int e = 0; e < 8; ++e) {
                int row = bm + wm * 32 + i * 16 + e + 8 * half;
                C[(size_t)row * N + col] = acc[i][j][e];
            }
        }
}

// ---------------------------------------------------------------------------
// RoPE + repack:  qkv f32 [b,l, 3*1024]  ->  Q,K f16 [b,h,l,dh] ; V^T f16 [b,h,dh,l]
// Softmax scale 1/sqrt(64) folded into Q.
// ---------------------------------------------------------------------------
__global__ void rope_pack(const float* __restrict__ qkv,
                          const float* __restrict__ cosT, const float* __restrict__ sinT,
                          _Float16* __restrict__ Q, _Float16* __restrict__ Kd,
                          _Float16* __restrict__ Vt) {
    int idx = blockIdx.x * blockDim.x + threadIdx.x;      // 2^22 threads
    int dh = idx & 63;
    int h  = (idx >> 6) & (NUM_HEADS - 1);
    int l  = (idx >> 10) & (SEQ - 1);
    int b  = idx >> 21;

    size_t base = ((size_t)b * SEQ + l) * (3 * D_MODEL);
    int e  = h * HEAD_DIM + dh;
    int ep = h * HEAD_DIM + ((dh + 32) & 63);

    float c = cosT[l * HEAD_DIM + dh];
    float s = sinT[l * HEAD_DIM + dh];

    float q0 = qkv[base + e],               qp = qkv[base + ep];
    float k0 = qkv[base + D_MODEL + e],     kp = qkv[base + D_MODEL + ep];
    float v0 = qkv[base + 2 * D_MODEL + e];

    float rq = (dh < 32) ? -qp : qp;
    float rk = (dh < 32) ? -kp : kp;
    float qr = (q0 * c + rq * s) * 0.125f;  // 1/sqrt(HEAD_DIM)
    float kr = (k0 * c + rk * s);

    int bh = b * NUM_HEADS + h;
    Q [((size_t)bh * SEQ + l) * HEAD_DIM + dh] = (_Float16)qr;
    Kd[((size_t)bh * SEQ + l) * HEAD_DIM + dh] = (_Float16)kr;
    Vt[((size_t)bh * HEAD_DIM + dh) * SEQ + l] = (_Float16)v0;
}

// ---------------------------------------------------------------------------
// Flash attention: block = one (b,h) x 128 queries; 8 waves x 16-query strips.
// Keys streamed in chunks of 64; online softmax in fp32 (one reduction pass
// per 64 keys instead of per 32 -> half the shuffle/barrier overhead).
// Output f16 [b*L + q][h*64 + dh]  (A-operand for the final projection GEMM).
// ---------------------------------------------------------------------------
__global__ __launch_bounds__(256)
void attn_f16(const _Float16* __restrict__ Q, const _Float16* __restrict__ Kd,
              const _Float16* __restrict__ Vt, _Float16* __restrict__ O) {
    __shared__ __align__(16) _Float16 Kblk[64][64];    // [key][dh]
    __shared__ __align__(16) _Float16 Vblk[64][64];    // [dh][key]
    __shared__ __align__(16) _Float16 Pbuf[8][16][64]; // per-wave C->A bounce

    const int t    = threadIdx.x;
    const int wave = t >> 5, lane = t & 31;
    const int half = lane >> 4, l16 = lane & 15;
    const int bh    = blockIdx.x;          // 0..31
    const int qtile = blockIdx.y;          // 0..15
    const int qbase = qtile * 128 + wave * 16;

    // Q strip: 16 queries x 64 dh = two 16x32 A fragments, loaded straight from global
    Frag16 qf[2];
#pragma unroll
    for (int kk = 0; kk < 2; ++kk) {
        const _Float16* qp = Q + ((size_t)bh * SEQ + qbase + l16) * HEAD_DIM + kk * 32;
        qf[kk].u[0] = *(const uint4*)(qp + half * 8);
        qf[kk].u[1] = *(const uint4*)(qp + 16 + half * 8);
    }

    v8f o[4] = {};
    float mrun[8], lrun[8];
#pragma unroll
    for (int e = 0; e < 8; ++e) { mrun[e] = -1e30f; lrun[e] = 0.f; }

    const int lr = t >> 2, lc = (t & 3) * 16;   // 64x64 tile loads: 16 elems/thread

    for (int kb = 0; kb < SEQ; kb += 64) {
        __syncthreads();
        // K block 64x64 [key][dh]
        *(uint4*)&Kblk[lr][lc]     = *(const uint4*)&Kd[((size_t)bh * SEQ + kb + lr) * HEAD_DIM + lc];
        *(uint4*)&Kblk[lr][lc + 8] = *(const uint4*)&Kd[((size_t)bh * SEQ + kb + lr) * HEAD_DIM + lc + 8];
        // V block 64x64 [dh][key] from V^T
        *(uint4*)&Vblk[lr][lc]     = *(const uint4*)&Vt[((size_t)bh * HEAD_DIM + lr) * SEQ + kb + lc];
        *(uint4*)&Vblk[lr][lc + 8] = *(const uint4*)&Vt[((size_t)bh * HEAD_DIM + lr) * SEQ + kb + lc + 8];
        __syncthreads();

        // S = Q * K^T : 16 queries x 64 keys (4 N-fragments, K-dim 64 = 2 steps)
        v8f s[4] = {};
#pragma unroll
        for (int nf = 0; nf < 4; ++nf)
#pragma unroll
            for (int kk = 0; kk < 2; ++kk) {
                Frag16 b;
                int kr = nf * 16 + l16;
                b.u[0] = *(const uint4*)&Kblk[kr][kk * 32 + half * 8];
                b.u[1] = *(const uint4*)&Kblk[kr][kk * 32 + 16 + half * 8];
                s[nf] = __builtin_amdgcn_wmma_f32_16x16x32_f16(
                    false, qf[kk].v, false, b.v, (short)0, s[nf], false, false);
            }

        // --- online softmax (rows live across the 16 lanes of each half-wave) ---
        float mcur[8];
#pragma unroll
        for (int e = 0; e < 8; ++e)
            mcur[e] = fmaxf(fmaxf(s[0][e], s[1][e]), fmaxf(s[2][e], s[3][e]));
#pragma unroll
        for (int msk = 1; msk < 16; msk <<= 1)
#pragma unroll
            for (int e = 0; e < 8; ++e)
                mcur[e] = fmaxf(mcur[e], __shfl_xor(mcur[e], msk, 32));

        float alpha[8];
#pragma unroll
        for (int e = 0; e < 8; ++e) {
            float mn = fmaxf(mrun[e], mcur[e]);
            alpha[e] = __expf(mrun[e] - mn);
            mrun[e]  = mn;
        }
#pragma unroll
        for (int nf = 0; nf < 4; ++nf)
#pragma unroll
            for (int e = 0; e < 8; ++e) s[nf][e] = __expf(s[nf][e] - mrun[e]);

        float rsum[8];
#pragma unroll
        for (int e = 0; e < 8; ++e)
            rsum[e] = (s[0][e] + s[1][e]) + (s[2][e] + s[3][e]);
#pragma unroll
        for (int msk = 1; msk < 16; msk <<= 1)
#pragma unroll
            for (int e = 0; e < 8; ++e) rsum[e] += __shfl_xor(rsum[e], msk, 32);
#pragma unroll
        for (int e = 0; e < 8; ++e) lrun[e] = lrun[e] * alpha[e] + rsum[e];
#pragma unroll
        for (int jf = 0; jf < 4; ++jf)
#pragma unroll
            for (int e = 0; e < 8; ++e) o[jf][e] *= alpha[e];

        // P: C-layout -> A-layout via per-wave LDS bounce (DS ops are in-order per wave)
#pragma unroll
        for (int nf = 0; nf < 4; ++nf)
#pragma unroll
            for (int e = 0; e < 8; ++e)
                Pbuf[wave][e + 8 * half][nf * 16 + l16] = (_Float16)s[nf][e];

        Frag16 p[2];
#pragma unroll
        for (int kk = 0; kk < 2; ++kk) {
            p[kk].u[0] = *(const uint4*)&Pbuf[wave][l16][kk * 32 + half * 8];
            p[kk].u[1] = *(const uint4*)&Pbuf[wave][l16][kk * 32 + 16 + half * 8];
        }

        // O += P * V : 16 queries x 64 dh (4 N-fragments, K=64 keys = 2 steps)
#pragma unroll
        for (int jf = 0; jf < 4; ++jf)
#pragma unroll
            for (int kk = 0; kk < 2; ++kk) {
                Frag16 b;
                int dh = jf * 16 + l16;
                b.u[0] = *(const uint4*)&Vblk[dh][kk * 32 + half * 8];
                b.u[1] = *(const uint4*)&Vblk[dh][kk * 32 + 16 + half * 8];
                o[jf] = __builtin_amdgcn_wmma_f32_16x16x32_f16(
                    false, p[kk].v, false, b.v, (short)0, o[jf], false, false);
            }
    }

    // normalize + store f16 [b*L + q][h*64 + dh]
    const int b = bh >> 4, h = bh & (NUM_HEADS - 1);
#pragma unroll
    for (int jf = 0; jf < 4; ++jf) {
        int col = h * HEAD_DIM + jf * 16 + l16;
#pragma unroll
        for (int e = 0; e < 8; ++e) {
            int qrow = qbase + e + 8 * half;
            size_t m = (size_t)b * SEQ + qrow;
            O[m * D_MODEL + col] = (_Float16)(o[jf][e] / lrun[e]);
        }
    }
}

// ---------------------------------------------------------------------------
// Launcher
// ---------------------------------------------------------------------------
extern "C" void kernel_launch(void* const* d_in, const int* in_sizes, int n_in,
                              void* d_out, int out_size, void* d_ws, size_t ws_size,
                              hipStream_t stream) {
    (void)in_sizes; (void)n_in; (void)out_size; (void)ws_size;

    const float* x    = (const float*)d_in[0];   // [2,2048,1024]
    const float* cosT = (const float*)d_in[1];   // [2048,64]
    const float* sinT = (const float*)d_in[2];   // [2048,64]
    const float* wqkv = (const float*)d_in[3];   // [3072,1024]
    const float* wo   = (const float*)d_in[4];   // [1024,1024]
    float* out = (float*)d_out;                  // [2,2048,1024]

    char* ws = (char*)d_ws;
    _Float16* x_h  = (_Float16*)(ws);                        //  8 MB
    _Float16* wq_h = (_Float16*)(ws + ( 8ull << 20));        //  6 MB
    _Float16* wo_h = (_Float16*)(ws + (14ull << 20));        //  2 MB
    float*    qkv  = (float*)   (ws + (16ull << 20));        // 48 MB
    _Float16* q_h  = (_Float16*)(ws + (64ull << 20));        //  8 MB
    _Float16* k_h  = (_Float16*)(ws + (72ull << 20));        //  8 MB
    _Float16* vt_h = (_Float16*)(ws + (80ull << 20));        //  8 MB
    _Float16* at_h = (_Float16*)(ws + (88ull << 20));        //  8 MB

    cvt_f32_f16<<<(MROWS * D_MODEL) / 256, 256, 0, stream>>>(x, x_h, MROWS * D_MODEL);
    cvt_f32_f16<<<(3 * D_MODEL * D_MODEL) / 256, 256, 0, stream>>>(wqkv, wq_h, 3 * D_MODEL * D_MODEL);
    cvt_f32_f16<<<(D_MODEL * D_MODEL) / 256, 256, 0, stream>>>(wo, wo_h, D_MODEL * D_MODEL);

    // qkv = x @ w_qkv^T   : M=4096, N=3072, K=1024
    gemm_nt_f16<<<dim3(3072 / 128, MROWS / 128), 256, 0, stream>>>(x_h, wq_h, qkv, MROWS, 3072, 1024);

    rope_pack<<<(BATCH * SEQ * NUM_HEADS * HEAD_DIM) / 256, 256, 0, stream>>>(
        qkv, cosT, sinT, q_h, k_h, vt_h);

    attn_f16<<<dim3(BATCH * NUM_HEADS, SEQ / 128), 256, 0, stream>>>(q_h, k_h, vt_h, at_h);

    // out = attn @ w_o^T  : M=4096, N=1024, K=1024
    gemm_nt_f16<<<dim3(D_MODEL / 128, MROWS / 128), 256, 0, stream>>>(at_h, wo_h, out, MROWS, D_MODEL, 1024);
}
